// WaveletMambaBranch_69363721830792
// MI455X (gfx1250) — matrix operand order repaired and checked by reading, not compiled, not run on hardware
//
#include <hip/hip_runtime.h>
#include <cstdint>
#include <cstddef>

// ---------------------------------------------------------------------------
// WaveletMambaBranch for MI455X (gfx1250, wave32, WMMA).
// Heavy GEMMs: v_wmma_f32_16x16x32_f16 (f16 A/B, f32 accum).
// Operand packs use 128-bit LDS loads (A row-major, B stored transposed).
// Global staging: batched float4 loads (single wait), then convert+store.
// ---------------------------------------------------------------------------

typedef __attribute__((ext_vector_type(16))) _Float16 v16h;
typedef __attribute__((ext_vector_type(8)))  _Float16 v8h;
typedef __attribute__((ext_vector_type(8)))  float    v8f;

#define WMMA16x16x32(a, b, c) \
  __builtin_amdgcn_wmma_f32_16x16x32_f16(false, (a), false, (b), (short)0, (c), false, false)

// db4 decomposition filters (reference order)
__constant__ float c_db4_lo[8] = {
  -0.010597401785069032f,  0.0328830116668852f,
   0.030841381835560764f, -0.18703481171909309f,
  -0.027983769416859854f,  0.6308807679298589f,
   0.7148465705529157f,    0.2303778133088965f };
__constant__ float c_db4_hi[8] = {
  -0.2303778133088965f,    0.7148465705529157f,
  -0.6308807679298589f,   -0.027983769416859854f,
   0.18703481171909309f,   0.030841381835560764f,
  -0.0328830116668852f,   -0.010597401785069032f };

// -------------------- WMMA operand packing (CDNA5 wave32 layouts) ----------
// A (16x32 f16), row-major LDS tile, stride in halves (multiple of 8):
// lane<16: row M=lane, K={half*8+0..7, 16+half*8+0..7} -> two contiguous 16B runs
__device__ __forceinline__ v16h pack_A(const _Float16* As, int stride, int lane) {
  int half = lane >> 4, m = lane & 15;
  const _Float16* p = As + m * stride + half * 8;
  v8h lo = *(const v8h*)(p);        // ds_load_b128
  v8h hi = *(const v8h*)(p + 16);   // ds_load_b128
  return __builtin_shufflevector(lo, hi, 0,1,2,3,4,5,6,7,8,9,10,11,12,13,14,15);
}
// B (32x16 f16) from a TRANSPOSED LDS tile Bt[n][k] (kstride multiple of 8):
// lane holds column n=lane&15, K={half*8+0..7, 16+half*8+0..7} -> two contiguous 16B runs
__device__ __forceinline__ v16h pack_Bt(const _Float16* Bt, int kstride, int lane) {
  int n = lane & 15, kb = (lane >> 4) * 8;
  const _Float16* p = Bt + n * kstride + kb;
  v8h lo = *(const v8h*)(p);
  v8h hi = *(const v8h*)(p + 16);
  return __builtin_shufflevector(lo, hi, 0,1,2,3,4,5,6,7,8,9,10,11,12,13,14,15);
}

__device__ __forceinline__ float silu_f(float x)     { return x / (1.f + __expf(-x)); }
__device__ __forceinline__ float softplus_f(float x) { return (x > 20.f) ? x : log1pf(__expf(x)); }
__device__ __forceinline__ float gelu_f(float x)     { return 0.5f * x * (1.f + erff(x * 0.70710678118654752f)); }

// -------------------- Kernel 1: 2-level db4 decomposition ------------------
__global__ __launch_bounds__(256) void wavelet_kernel(const float* __restrict__ x,
                                                      float* __restrict__ comps) {
  int b = blockIdx.x >> 5, c = blockIdx.x & 31;
  __shared__ float row[512];
  __shared__ float lo[512];
  for (int l = threadIdx.x; l < 512; l += 256)
    row[l] = x[((size_t)b * 512 + l) * 32 + c];
  __syncthreads();
  size_t base = (size_t)(b * 32 + c) * 512;
  float* hi1 = comps;
  float* hi2 = comps + 262144;
  float* lo2 = comps + 524288;
  for (int l = threadIdx.x; l < 512; l += 256) {
    float h = 0.f, lv = 0.f;
#pragma unroll
    for (int k = 0; k < 8; ++k) {
      int m = l + k - 7; if (m < 0) m = -m;      // reflect pad
      float xv = row[m];
      h  += c_db4_hi[k] * xv;
      lv += c_db4_lo[k] * xv;
    }
    hi1[base + l] = h;
    lo[l] = lv;
  }
  __syncthreads();
  for (int l = threadIdx.x; l < 512; l += 256) {
    float h = 0.f, lv = 0.f;
#pragma unroll
    for (int k = 0; k < 8; ++k) {
      int m = l + k - 7; if (m < 0) m = -m;
      float xv = lo[m];
      h  += c_db4_hi[k] * xv;
      lv += c_db4_lo[k] * xv;
    }
    hi2[base + l] = h;
    lo2[base + l] = lv;
  }
}

// -------------------- Kernel 2: patch embed (tiny K, VALU) -----------------
__global__ __launch_bounds__(256) void pe_proj_kernel(const float* __restrict__ comp,
                                                      const float* __restrict__ pe_w,
                                                      const float* __restrict__ pe_b,
                                                      float* __restrict__ X0,
                                                      int N, int psz) {
  size_t idx = (size_t)blockIdx.x * 256 + threadIdx.x;   // over T*128
  size_t total = (size_t)16 * N * 32 * 128;
  if (idx >= total) return;
  int d = (int)(idx & 127);
  size_t t = idx >> 7;
  int c = (int)(t & 31);
  size_t bn = t >> 5;
  int n = (int)(bn % N), b = (int)(bn / N);
  const float* src = comp + (size_t)(b * 32 + c) * 512 + n * psz;
  float acc = pe_b[d];
  for (int j = 0; j < psz; ++j) acc += src[j] * pe_w[j * 128 + d];
  X0[idx] = acc;
}

// -------------------- Kernel 3: LayerNorm (one row per wave32) -------------
__global__ __launch_bounds__(256) void layernorm_kernel(const float* __restrict__ in,
                                                        const float* __restrict__ g,
                                                        const float* __restrict__ bta,
                                                        float* __restrict__ out, int rows) {
  int row  = (int)(((size_t)blockIdx.x * 256 + threadIdx.x) >> 5);
  int lane = threadIdx.x & 31;
  if (row >= rows) return;
  const float* xp = in + (size_t)row * 128;
  float v[4], s = 0.f, ss = 0.f;
#pragma unroll
  for (int i = 0; i < 4; ++i) {
    v[i] = xp[lane + 32 * i];
    s += v[i]; ss += v[i] * v[i];
  }
#pragma unroll
  for (int off = 16; off; off >>= 1) {
    s  += __shfl_xor(s,  off, 32);
    ss += __shfl_xor(ss, off, 32);
  }
  float mean = s * (1.f / 128.f);
  float var  = ss * (1.f / 128.f) - mean * mean;
  float r = rsqrtf(var + 1e-5f);
  float* op = out + (size_t)row * 128;
#pragma unroll
  for (int i = 0; i < 4; ++i) {
    int d = lane + 32 * i;
    op[d] = (v[i] - mean) * r * g[d] + bta[d];
  }
}

// -------------------- Kernel 4: fused token-mix MLP (WMMA) -----------------
template <int N>
__global__ __launch_bounds__(256) void tokenmix_kernel(const float* __restrict__ XN,
                                                       const float* __restrict__ X0,
                                                       const float* __restrict__ t1_w,
                                                       const float* __restrict__ t1_b,
                                                       const float* __restrict__ t2_w,
                                                       const float* __restrict__ t2_b,
                                                       float* __restrict__ X1) {
  constexpr int H     = 2 * N;
  constexpr int TPW1  = H / 64;             // GEMM1 tiles per wave: 4/2/1
  constexpr int TOT2  = N / 8;              // GEMM2 output tiles: 16/8/4
  constexpr int TPW2  = (TOT2 + 7) / 8;     // 2/1/1
  constexpr int ATRIP = (N * 8) / 256;      // 4/2/1 float4 loads per thread
  constexpr int W1TRIP = (8 * H) / 256;     // 8/4/2
  constexpr int W2TRIP = (8 * N) / 256;     // 4/2/1
  __shared__ alignas(16) _Float16 As[32 * 128];   // input tile  [32][N], stride 128
  __shared__ alignas(16) _Float16 Hs[32 * 256];   // hidden tile [32][H], stride 256
  __shared__ alignas(16) _Float16 Wt[256 * 32];   // weight K-chunk, TRANSPOSED [n][k]
  int tid = threadIdx.x, lane = tid & 31, wave = tid >> 5;
  int r0 = blockIdx.x * 32;
  int bcIdx = r0 >> 7, dfeat0 = r0 & 127;
  int b = bcIdx >> 5, c = bcIdx & 31;

  // A tile: batched float4 loads along contiguous dfeat axis, then store As[rl][n]
  {
    float4 av[ATRIP];
#pragma unroll
    for (int u = 0; u < ATRIP; ++u) {
      int i = tid + u * 256;
      int n = i >> 3, rl = (i & 7) * 4;
      av[u] = *(const float4*)(XN + ((size_t)(b * N + n) * 32 + c) * 128 + dfeat0 + rl);
    }
#pragma unroll
    for (int u = 0; u < ATRIP; ++u) {
      int i = tid + u * 256;
      int n = i >> 3, rl = (i & 7) * 4;
      As[(rl + 0) * 128 + n] = (_Float16)av[u].x;
      As[(rl + 1) * 128 + n] = (_Float16)av[u].y;
      As[(rl + 2) * 128 + n] = (_Float16)av[u].z;
      As[(rl + 3) * 128 + n] = (_Float16)av[u].w;
    }
  }

  // ---- GEMM1: [32,N] @ [N,H] -> hidden ----
  v8f acc1[TPW1] = {};
  for (int k0 = 0; k0 < N; k0 += 32) {
    {
      float4 wv[W1TRIP];
#pragma unroll
      for (int u = 0; u < W1TRIP; ++u) {
        int i = tid + u * 256;
        int rr = i / (H / 4), cc = (i % (H / 4)) * 4;
        wv[u] = *(const float4*)(t1_w + (size_t)(k0 + rr) * H + cc);
      }
#pragma unroll
      for (int u = 0; u < W1TRIP; ++u) {
        int i = tid + u * 256;
        int rr = i / (H / 4), cc = (i % (H / 4)) * 4;
        Wt[(cc + 0) * 32 + rr] = (_Float16)wv[u].x;
        Wt[(cc + 1) * 32 + rr] = (_Float16)wv[u].y;
        Wt[(cc + 2) * 32 + rr] = (_Float16)wv[u].z;
        Wt[(cc + 3) * 32 + rr] = (_Float16)wv[u].w;
      }
    }
    __syncthreads();
#pragma unroll
    for (int ti = 0; ti < TPW1; ++ti) {
      int t = wave + ti * 8;
      int msub = t & 1, col = t >> 1;
      v16h a  = pack_A(&As[(msub * 16) * 128 + k0], 128, lane);
      v16h bb = pack_Bt(&Wt[(col * 16) * 32], 32, lane);
      acc1[ti] = WMMA16x16x32(a, bb, acc1[ti]);
    }
    __syncthreads();
  }
  // bias + gelu -> Hs (f16)
  {
    int nl = lane & 15, mb = (lane >> 4) * 8;
#pragma unroll
    for (int ti = 0; ti < TPW1; ++ti) {
      int t = wave + ti * 8;
      int msub = t & 1, col = t >> 1;
      int gcol = col * 16 + nl;
      float bv = t1_b[gcol];
#pragma unroll
      for (int j = 0; j < 8; ++j) {
        float xv = acc1[ti][j] + bv;
        Hs[(msub * 16 + mb + j) * 256 + gcol] = (_Float16)gelu_f(xv);
      }
    }
  }
  __syncthreads();

  // ---- GEMM2: [32,H] @ [H,N] -> mix ----
  v8f acc2[TPW2] = {};
  for (int k0 = 0; k0 < H; k0 += 32) {
    {
      float4 wv[W2TRIP];
#pragma unroll
      for (int u = 0; u < W2TRIP; ++u) {
        int i = tid + u * 256;
        int rr = i / (N / 4), cc = (i % (N / 4)) * 4;
        wv[u] = *(const float4*)(t2_w + (size_t)(k0 + rr) * N + cc);
      }
#pragma unroll
      for (int u = 0; u < W2TRIP; ++u) {
        int i = tid + u * 256;
        int rr = i / (N / 4), cc = (i % (N / 4)) * 4;
        Wt[(cc + 0) * 32 + rr] = (_Float16)wv[u].x;
        Wt[(cc + 1) * 32 + rr] = (_Float16)wv[u].y;
        Wt[(cc + 2) * 32 + rr] = (_Float16)wv[u].z;
        Wt[(cc + 3) * 32 + rr] = (_Float16)wv[u].w;
      }
    }
    __syncthreads();
#pragma unroll
    for (int ti = 0; ti < TPW2; ++ti) {
      int t = wave + ti * 8;
      if (t < TOT2) {                      // wave-uniform guard (EXEC all-1 inside)
        int msub = t & 1, col = t >> 1;
        v16h a  = pack_A(&Hs[(msub * 16) * 256 + k0], 256, lane);
        v16h bb = pack_Bt(&Wt[(col * 16) * 32], 32, lane);
        acc2[ti] = WMMA16x16x32(a, bb, acc2[ti]);
      }
    }
    __syncthreads();
  }
  // bias + residual -> X1 (8 contiguous floats per lane: two float4s)
  {
    int nl = lane & 15, mb = (lane >> 4) * 8;
#pragma unroll
    for (int ti = 0; ti < TPW2; ++ti) {
      int t = wave + ti * 8;
      if (t < TOT2) {
        int msub = t & 1, col = t >> 1;
        int gn = col * 16 + nl;
        float bv = t2_b[gn];
        size_t basea = ((size_t)(b * N + gn) * 32 + c) * 128 + dfeat0 + msub * 16 + mb;
        float4 r0 = *(const float4*)(X0 + basea);
        float4 r1 = *(const float4*)(X0 + basea + 4);
        float4 o0, o1;
        o0.x = r0.x + acc2[ti][0] + bv;  o0.y = r0.y + acc2[ti][1] + bv;
        o0.z = r0.z + acc2[ti][2] + bv;  o0.w = r0.w + acc2[ti][3] + bv;
        o1.x = r1.x + acc2[ti][4] + bv;  o1.y = r1.y + acc2[ti][5] + bv;
        o1.z = r1.z + acc2[ti][6] + bv;  o1.w = r1.w + acc2[ti][7] + bv;
        *(float4*)(X1 + basea)     = o0;
        *(float4*)(X1 + basea + 4) = o1;
      }
    }
  }
}

// -------------------- Kernel 5: generic WMMA GEMM (64x64 tile) -------------
// C[M,N] = A[M,K] @ B[K,N] (+bias[n]) (+resid[M,N])
// Requires: M % 64 == 0, K % 32 == 0, lda/ldb % 4 == 0.
// BOUND_N=true handles N not multiple of 64 via clamped loads + guarded stores.
template <bool BOUND_N>
__global__ __launch_bounds__(256) void gemm_wmma_kernel(const float* __restrict__ A, int lda,
                                                        const float* __restrict__ Bw, int ldb,
                                                        const float* __restrict__ bias,
                                                        const float* __restrict__ resid,
                                                        float* __restrict__ C, int ldc,
                                                        int N, int K) {
  __shared__ alignas(16) _Float16 As[64 * 32];    // [m][k]
  __shared__ alignas(16) _Float16 Bt[64 * 32];    // TRANSPOSED [n][k]
  int tid = threadIdx.x, lane = tid & 31, wave = tid >> 5;
  int m0 = blockIdx.x * 64;
  int n0 = blockIdx.y * 64;
  int wm  = (wave & 3) * 16;
  int wnb = (wave >> 2) * 16;          // this wave's col groups: wnb, wnb+32
  v8f acc0 = {}, acc1 = {};
  for (int k0 = 0; k0 < K; k0 += 32) {
    // ---- batched load phase: 4 global b128 loads in flight ----
    float4 av[2], bv[2][4];
#pragma unroll
    for (int u = 0; u < 2; ++u) {      // A: 64x32 floats = 512 float4
      int i = tid + u * 256;
      int r = i >> 3, c4 = (i & 7) * 4;
      av[u] = *(const float4*)(A + (size_t)(m0 + r) * lda + k0 + c4);
    }
#pragma unroll
    for (int u = 0; u < 2; ++u) {      // B: 32x64 floats = 512 float4
      int i = tid + u * 256;
      int r = i >> 4, c4 = (i & 15) * 4;   // r = k-local, c4 = n-local
      if (!BOUND_N) {
        const float4 t = *(const float4*)(Bw + (size_t)(k0 + r) * ldb + n0 + c4);
        bv[u][0] = t; bv[u][1] = t; bv[u][2] = t; bv[u][3] = t;   // lanes use .x/.y/.z/.w
      } else {
#pragma unroll
        for (int e = 0; e < 4; ++e) {
          int gn = n0 + c4 + e;
          int gnc = (gn < N) ? gn : (N - 1);  // clamp; cols>=N discarded at store
          bv[u][e].x = Bw[(size_t)(k0 + r) * ldb + gnc];
        }
      }
    }
    // ---- convert + LDS store phase ----
#pragma unroll
    for (int u = 0; u < 2; ++u) {
      int i = tid + u * 256;
      int r = i >> 3, c4 = (i & 7) * 4;
      As[r * 32 + c4 + 0] = (_Float16)av[u].x;
      As[r * 32 + c4 + 1] = (_Float16)av[u].y;
      As[r * 32 + c4 + 2] = (_Float16)av[u].z;
      As[r * 32 + c4 + 3] = (_Float16)av[u].w;
    }
#pragma unroll
    for (int u = 0; u < 2; ++u) {
      int i = tid + u * 256;
      int r = i >> 4, c4 = (i & 15) * 4;
      float e0, e1, e2, e3;
      if (!BOUND_N) { e0 = bv[u][0].x; e1 = bv[u][1].y; e2 = bv[u][2].z; e3 = bv[u][3].w; }
      else          { e0 = bv[u][0].x; e1 = bv[u][1].x; e2 = bv[u][2].x; e3 = bv[u][3].x; }
      Bt[(c4 + 0) * 32 + r] = (_Float16)e0;
      Bt[(c4 + 1) * 32 + r] = (_Float16)e1;
      Bt[(c4 + 2) * 32 + r] = (_Float16)e2;
      Bt[(c4 + 3) * 32 + r] = (_Float16)e3;
    }
    if (k0 + 32 < K)   // prefetch next weight chunk -> global_prefetch_b8
      __builtin_prefetch((const void*)&Bw[(size_t)(k0 + 32) * ldb + n0 + (tid & 63)], 0, 1);
    __syncthreads();
    v16h a  = pack_A(&As[wm * 32], 32, lane);
    v16h b0 = pack_Bt(&Bt[wnb * 32], 32, lane);
    v16h b1 = pack_Bt(&Bt[(wnb + 32) * 32], 32, lane);
    acc0 = WMMA16x16x32(a, b0, acc0);
    acc1 = WMMA16x16x32(a, b1, acc1);
    __syncthreads();
  }
  int nl = lane & 15, mb = (lane >> 4) * 8;
#pragma unroll
  for (int g = 0; g < 2; ++g) {
    const v8f& acc = g ? acc1 : acc0;
    int gn = n0 + wnb + g * 32 + nl;
    if (!BOUND_N || gn < N) {
      float bv2 = bias ? bias[gn] : 0.f;
#pragma unroll
      for (int j = 0; j < 8; ++j) {
        int gm = m0 + wm + mb + j;
        float v = acc[j] + bv2;
        if (resid) v += resid[(size_t)gm * ldc + gn];
        C[(size_t)gm * ldc + gn] = v;
      }
    }
  }
}

// -------------------- Kernel 6: causal depthwise conv + SiLU ---------------
__global__ __launch_bounds__(256) void conv_silu_kernel(const float* __restrict__ XZ,
                                                        const float* __restrict__ conv_w,
                                                        const float* __restrict__ conv_b,
                                                        float* __restrict__ XC, int T) {
  size_t idx = (size_t)blockIdx.x * 256 + threadIdx.x;   // T*256
  if (idx >= (size_t)T * 256) return;
  int ch = (int)(idx & 255);
  size_t t = idx >> 8;
  int l = (int)(t & 31);
  size_t bn = t >> 5;
  float acc = conv_b[ch];
#pragma unroll
  for (int k = 0; k < 4; ++k) {
    int ll = l + k - 3;
    if (ll >= 0) acc += conv_w[ch * 4 + k] * XZ[((bn << 5) + ll) * 512 + ch];
  }
  XC[idx] = silu_f(acc);
}

// -------------------- Kernel 6b: dt projection (K=8, VALU) -----------------
// DELTA[t,ch] = dt_b[ch] + sum_{j<8} DBL[t,j] * dt_w[j,ch]   (pre-softplus)
__global__ __launch_bounds__(256) void dtproj_kernel(const float* __restrict__ DBL,
                                                     const float* __restrict__ dt_w,
                                                     const float* __restrict__ dt_b,
                                                     float* __restrict__ DELTA, int T) {
  size_t idx = (size_t)blockIdx.x * 256 + threadIdx.x;   // T*256
  if (idx >= (size_t)T * 256) return;
  int ch = (int)(idx & 255);
  size_t t = idx >> 8;
  float acc = dt_b[ch];
#pragma unroll
  for (int j = 0; j < 8; ++j) acc += DBL[t * 40 + j] * dt_w[j * 256 + ch];
  DELTA[idx] = acc;
}

// -------------------- Kernel 7: selective scan + gating --------------------
__global__ __launch_bounds__(256) void scan_kernel(const float* __restrict__ DELTAraw,
                                                   const float* __restrict__ DBL,
                                                   const float* __restrict__ XZ,
                                                   const float* __restrict__ A_log,
                                                   const float* __restrict__ Dp,
                                                   float* __restrict__ Y) {
  int bn = blockIdx.x, ch = threadIdx.x;
  __shared__ float Bs[32][16];
  __shared__ float Cs[32][16];
  for (int i = threadIdx.x; i < 512; i += 256) {
    int l = i >> 4, s = i & 15;
    size_t t = (size_t)bn * 32 + l;
    Bs[l][s] = DBL[t * 40 + 8 + s];
    Cs[l][s] = DBL[t * 40 + 24 + s];
  }
  __syncthreads();
  float Arow[16];
#pragma unroll
  for (int s = 0; s < 16; ++s) Arow[s] = -__expf(A_log[ch * 16 + s]);
  float Dv = Dp[ch];
  float h[16];
#pragma unroll
  for (int s = 0; s < 16; ++s) h[s] = 0.f;
  for (int l = 0; l < 32; ++l) {
    size_t t = (size_t)bn * 32 + l;
    float u  = Y[t * 256 + ch];
    float dt = softplus_f(DELTAraw[t * 256 + ch]);
    float du = dt * u;
    float y = 0.f;
#pragma unroll
    for (int s = 0; s < 16; ++s) {
      h[s] = h[s] * __expf(dt * Arow[s]) + du * Bs[l][s];
      y += h[s] * Cs[l][s];
    }
    y += u * Dv;
    float z = XZ[t * 512 + 256 + ch];
    Y[t * 256 + ch] = y * silu_f(z);
  }
}

// ---------------------------------------------------------------------------
extern "C" void kernel_launch(void* const* d_in, const int* in_sizes, int n_in,
                              void* d_out, int out_size, void* d_ws, size_t ws_size,
                              hipStream_t stream) {
  (void)in_sizes; (void)n_in; (void)out_size; (void)ws_size;
  const float* x = (const float*)d_in[0];
  float* out = (float*)d_out;
  float* ws  = (float*)d_ws;

  // workspace layout (floats); sized for branch 0; ~302 MB total
  float* comps = ws;                          // 3 * 262144
  float* bufA  = ws + 786432;                 // X0 / residual1      (8388608)
  float* bufB  = bufA + 8388608;              // LN scratch          (8388608)
  float* DELTA = bufA;                        // alias A+B (16777216) — safe: A,B dead
  float* bufC  = bufB + 8388608;              // X1 / residual2      (8388608)
  float* XZ    = bufC + 8388608;              // in_proj out         (33554432)
  float* XC    = XZ + 33554432;               // conv out / y        (16777216)
  float* DBL   = XC + 16777216;               // x_proj out          (2621440)

  wavelet_kernel<<<512, 256, 0, stream>>>(x, comps);

  static const int PSZ[3] = {4, 8, 16};
  static const int NN[3]  = {128, 64, 32};
  static const size_t OUT_OFF[3] = {0, 8388608, 12582912};

  for (int i = 0; i < 3; ++i) {
    const int base = 1 + i * 19;
    const float* pe_w    = (const float*)d_in[base + 0];
    const float* pe_b    = (const float*)d_in[base + 1];
    const float* ln1_g   = (const float*)d_in[base + 2];
    const float* ln1_b   = (const float*)d_in[base + 3];
    const float* t1_w    = (const float*)d_in[base + 4];
    const float* t1_b    = (const float*)d_in[base + 5];
    const float* t2_w    = (const float*)d_in[base + 6];
    const float* t2_b    = (const float*)d_in[base + 7];
    const float* ln2_g   = (const float*)d_in[base + 8];
    const float* ln2_b   = (const float*)d_in[base + 9];
    const float* in_w    = (const float*)d_in[base + 10];
    const float* conv_w  = (const float*)d_in[base + 11];
    const float* conv_b  = (const float*)d_in[base + 12];
    const float* xproj_w = (const float*)d_in[base + 13];
    const float* dt_w    = (const float*)d_in[base + 14];
    const float* dt_b    = (const float*)d_in[base + 15];
    const float* A_log   = (const float*)d_in[base + 16];
    const float* Dp      = (const float*)d_in[base + 17];
    const float* out_w   = (const float*)d_in[base + 18];

    const int N   = NN[i];
    const int psz = PSZ[i];
    const int BN  = 16 * N;
    const int T   = BN * 32;          // rows of [*,128] tokens (mult of 16384)

    // 1) patch embed -> bufA (X0)
    pe_proj_kernel<<<(T * 128) / 256, 256, 0, stream>>>(
        comps + (size_t)i * 262144, pe_w, pe_b, bufA, N, psz);
    // 2) LN1 -> bufB
    layernorm_kernel<<<T / 8, 256, 0, stream>>>(bufA, ln1_g, ln1_b, bufB, T);
    // 3) fused token MLP + residual -> bufC (X1)
    if (N == 128)
      tokenmix_kernel<128><<<2048, 256, 0, stream>>>(bufB, bufA, t1_w, t1_b, t2_w, t2_b, bufC);
    else if (N == 64)
      tokenmix_kernel<64><<<2048, 256, 0, stream>>>(bufB, bufA, t1_w, t1_b, t2_w, t2_b, bufC);
    else
      tokenmix_kernel<32><<<2048, 256, 0, stream>>>(bufB, bufA, t1_w, t1_b, t2_w, t2_b, bufC);
    // 4) LN2 -> bufB
    layernorm_kernel<<<T / 8, 256, 0, stream>>>(bufC, ln2_g, ln2_b, bufB, T);
    // 5) in_proj: [T,128]@[128,512] -> XZ
    gemm_wmma_kernel<false><<<dim3(T / 64, 8), 256, 0, stream>>>(
        bufB, 128, in_w, 512, nullptr, nullptr, XZ, 512, 512, 128);
    // 6) causal conv + SiLU -> XC
    conv_silu_kernel<<<T, 256, 0, stream>>>(XZ, conv_w, conv_b, XC, T);
    // 7) x_proj: [T,256]@[256,40] -> DBL  (N=40 -> bounded variant)
    gemm_wmma_kernel<true><<<dim3(T / 64, 1), 256, 0, stream>>>(
        XC, 256, xproj_w, 40, nullptr, nullptr, DBL, 40, 40, 256);
    // 8) dt proj (+bias, K=8): VALU kernel -> DELTA (pre-softplus)
    dtproj_kernel<<<T, 256, 0, stream>>>(DBL, dt_w, dt_b, DELTA, T);
    // 9) selective scan + gating (y written in place over XC)
    scan_kernel<<<BN, 256, 0, stream>>>(DELTA, DBL, XZ, A_log, Dp, XC);
    // 10) out_proj + residual -> d_out
    gemm_wmma_kernel<false><<<dim3(T / 64, 2), 256, 0, stream>>>(
        XC, 256, out_w, 128, nullptr, bufC, out + OUT_OFF[i], 128, 128, 256);
  }
}
